// VAE_4578435138216
// MI455X (gfx1250) — matrix operand order repaired and checked by reading, not compiled, not run on hardware
//
#include <hip/hip_runtime.h>
#include <cstdint>
#include <cstddef>

// ---------------------------------------------------------------------------
// Types for CDNA5 WMMA (wave32, v_wmma_f32_16x16x32_bf16, f32 accumulate)
// ---------------------------------------------------------------------------
typedef __attribute__((ext_vector_type(16))) __bf16 v16bf;
typedef __attribute__((ext_vector_type(8)))  float  v8f;

union AFrag { v16bf v; uint4 q[2]; };
union BFrag { v16bf v; unsigned int u[8]; };

__device__ __forceinline__ unsigned int f2bf(float f) {
    union { float f; unsigned int u; } x; x.f = f;
    unsigned int u = x.u;
    return (u + 0x7FFFu + ((u >> 16) & 1u)) >> 16;   // RNE fp32 -> bf16
}
__device__ __forceinline__ unsigned int pack2(float a, float b) {
    return f2bf(a) | (f2bf(b) << 16);
}

// ---------------------------------------------------------------------------
// Branch-free im2col B-fragment gather (bf16 activations).
// inbf is the UNIFORM kernel-arg pointer; all per-lane variation lives in the
// 32-bit offset so loads lower to scalar-base + 32-bit voffset form.
// okbits = per-lane 27-bit tap validity mask (halo), computed once.
// ---------------------------------------------------------------------------
template<bool TAIL>
__device__ __forceinline__ v16bf gatherB(const unsigned short* __restrict__ inbf,
                                         int ks, int K, unsigned okbits,
                                         int bBase, int Win, int HW, int DHW)
{
    int ci = ks / 27;
    int rem = ks - ci * 27;
    int kd = rem / 9; rem -= kd * 9;
    int kh = rem / 3; int kw = rem - kh * 3;
    int tap = kd * 9 + kh * 3 + kw;
    int off = bBase + ci * DHW + kd * HW + kh * Win + kw;
    BFrag f;
    #pragma unroll
    for (int i = 0; i < 16; ++i) {
        bool ok = ((okbits >> tap) & 1u) != 0u;
        if (TAIL) ok = ok && (ks + i < K);
        const int addr = ok ? off : 0;               // v_cndmask; load unconditional
        unsigned val = (unsigned)inbf[addr];
        val = ok ? val : 0u;                          // v_cndmask
        if (i & 1) f.u[i >> 1] |= val << 16; else f.u[i >> 1] = val;
        // advance tap walk (incremental adds only)
        ++tap; ++off; ++kw;
        if (kw == 3) { kw = 0; off += Win - 3;
            if (++kh == 3) { kh = 0; off += HW - 3 * Win;
                if (++kd == 3) { kd = 0; off += DHW - 3 * HW; tap = 0; } } }
    }
    return f.v;
}

// ---------------------------------------------------------------------------
// Implicit-GEMM 3D conv (pad=1, stride 1/2). M=Cout, N=B*spatial, K=Cin*27.
// One wave32 computes MT vertical 16x16 tiles, sharing the B fragment across
// MT WMMAs. B fragments are software-pipelined (double buffered) so the
// gather of step s+1 overlaps the WMMAs of step s and avoids the
// WMMA->VALU WAR hazard NOPs. Weights pre-packed bf16 [Cout][K32].
// ---------------------------------------------------------------------------
template<int MT>
__global__ void conv3_wmma_kernel(const unsigned short* __restrict__ inbf,
                                  const uint4* __restrict__ wpk,
                                  float* __restrict__ out,
                                  int Cin, int Cout, int B,
                                  int Din, int Hin, int Win,
                                  int Dout, int Hout, int Wout,
                                  int stride, int K, int K32, int nTiles)
{
    const int wave = blockIdx.x * (blockDim.x >> 5) + (threadIdx.x >> 5);
    if (wave >= nTiles) return;                 // wave-uniform
    const int lane = threadIdx.x & 31;
    const int half = lane >> 4;
    const int l16  = lane & 15;

    const int N = B * Dout * Hout * Wout;
    const int nTilesN = N >> 4;
    const int mTile = wave / nTilesN;
    const int nTile = wave - mTile * nTilesN;

    // ---- B column decode (N = nTile*16 + l16) ----
    int n = (nTile << 4) + l16;
    int ow = n % Wout; int t = n / Wout;
    int oh = t % Hout; t /= Hout;
    int od = t % Dout; const int b = t / Dout;
    const int idB = od * stride - 1;
    const int ihB = oh * stride - 1;
    const int iwB = ow * stride - 1;
    const int HW = Hin * Win, DHW = Din * HW;
    const int bBase = b * Cin * DHW + idB * HW + ihB * Win + iwB;  // 32-bit

    // per-lane halo validity bitmask over the 27 taps
    unsigned okbits = 0;
    #pragma unroll
    for (int kd = 0; kd < 3; ++kd) {
        const bool oD = (unsigned)(idB + kd) < (unsigned)Din;
        #pragma unroll
        for (int kh = 0; kh < 3; ++kh) {
            const bool oH = (unsigned)(ihB + kh) < (unsigned)Hin;
            #pragma unroll
            for (int kw = 0; kw < 3; ++kw) {
                const bool oW = (unsigned)(iwB + kw) < (unsigned)Win;
                if (oD && oH && oW) okbits |= 1u << (kd * 9 + kh * 3 + kw);
            }
        }
    }

    // ---- A row offsets (uint4 units, 32-bit) ----
    const int rowU4 = K32 >> 3;     // uint4 per weight row
    int wrowOff[MT];
    #pragma unroll
    for (int m = 0; m < MT; ++m)
        wrowOff[m] = ((mTile * MT + m) * 16 + l16) * rowU4;

    v8f acc[MT];
    #pragma unroll
    for (int m = 0; m < MT; ++m) acc[m] = v8f{0.f,0.f,0.f,0.f,0.f,0.f,0.f,0.f};

    const int nSteps  = K32 >> 5;
    const bool hasTail = (K & 31) != 0;

    auto gat = [&](int s) -> v16bf {
        const int ks = (s << 5) + 16 * half;
        if (hasTail && s == nSteps - 1)          // wave-uniform branch
            return gatherB<true >(inbf, ks, K, okbits, bBase, Win, HW, DHW);
        return gatherB<false>(inbf, ks, K, okbits, bBase, Win, HW, DHW);
    };

    v16bf Bcur = gat(0);
    for (int s = 0; s < nSteps; ++s) {
        v16bf Bnext = Bcur;
        if (s + 1 < nSteps) Bnext = gat(s + 1);  // prefetch overlaps WMMAs below
        const int k0 = s << 5;
        #pragma unroll
        for (int m = 0; m < MT; ++m) {
            const int idx = wrowOff[m] + (k0 >> 3) + 2 * half;
            AFrag A;
            A.q[0] = wpk[idx];
            A.q[1] = wpk[idx + 1];               // +16B: immediate offset form
            acc[m] = __builtin_amdgcn_wmma_f32_16x16x32_bf16(
                false, A.v, false, Bcur, (short)0, acc[m], false, false);
        }
        Bcur = Bnext;
    }

    // ---- C/D store: VGPR r -> M = r + 8*half, N = l16 (32-bit offsets) ----
    const int HWo = Hout * Wout;
    const int DoutHWo = Dout * HWo;
    const int outBase = b * Cout * DoutHWo + od * HWo + oh * Wout + ow;
    #pragma unroll
    for (int m = 0; m < MT; ++m) {
        const int coBase = (mTile * MT + m) * 16 + 8 * half;
        #pragma unroll
        for (int r = 0; r < 8; ++r)
            out[outBase + (coBase + r) * DoutHWo] = acc[m][r];
    }
}

// ---------------------------------------------------------------------------
// fp32 -> packed bf16 conversion (activations) and weight packing
// ---------------------------------------------------------------------------
__global__ void cvt_bf16_kernel(const float* __restrict__ x, unsigned* __restrict__ y, int nPairs)
{
    const int i = blockIdx.x * blockDim.x + threadIdx.x;
    if (i < nPairs) y[i] = pack2(x[2 * i], x[2 * i + 1]);
}

__global__ void pack_w_kernel(const float* __restrict__ w, unsigned* __restrict__ o,
                              int Cout, int K, int K32)
{
    const int pairs = K32 >> 1;
    const int i = blockIdx.x * blockDim.x + threadIdx.x;
    if (i >= Cout * pairs) return;
    const int co = i / pairs; const int k = 2 * (i - co * pairs);
    const float a0 = (k     < K) ? w[(size_t)co * K + k]     : 0.f;
    const float a1 = (k + 1 < K) ? w[(size_t)co * K + k + 1] : 0.f;
    o[i] = pack2(a0, a1);
}

// ---------------------------------------------------------------------------
// Elementwise / reduction helper kernels
// ---------------------------------------------------------------------------
__global__ void zero_kernel(float* p, int n) {
    int i = blockIdx.x * blockDim.x + threadIdx.x;
    if (i < n) p[i] = 0.f;
}

// stats layout: [0]=cnt, [1..512]=sum, [513..1024]=varsum
__global__ void bn_sum_kernel(const float* __restrict__ x, const float* __restrict__ mask,
                              float* __restrict__ stats, int C, int BS, int S)
{
    const int c = blockIdx.y;
    __shared__ float sh[256], shc[256];
    float loc = 0.f, locC = 0.f;
    for (int i = blockIdx.x * blockDim.x + threadIdx.x; i < BS; i += gridDim.x * blockDim.x) {
        const int b = i / S, s = i - b * S;
        const float m = mask ? mask[(size_t)b * S + s] : 1.f;
        loc += x[((size_t)b * C + c) * S + s] * m;
        if (c == 0) locC += m;
    }
    sh[threadIdx.x] = loc; shc[threadIdx.x] = locC; __syncthreads();
    for (int o = blockDim.x >> 1; o > 0; o >>= 1) {
        if (threadIdx.x < o) { sh[threadIdx.x] += sh[threadIdx.x + o]; shc[threadIdx.x] += shc[threadIdx.x + o]; }
        __syncthreads();
    }
    if (threadIdx.x == 0) {
        atomicAdd(&stats[1 + c], sh[0]);
        if (c == 0) atomicAdd(&stats[0], shc[0]);
    }
}

__global__ void bn_var_kernel(const float* __restrict__ x, const float* __restrict__ mask,
                              float* __restrict__ stats, int C, int BS, int S)
{
    const int c = blockIdx.y;
    const float cnt = fmaxf(stats[0], 1.f);
    const float mean = stats[1 + c] / cnt;
    __shared__ float sh[256];
    float loc = 0.f;
    for (int i = blockIdx.x * blockDim.x + threadIdx.x; i < BS; i += gridDim.x * blockDim.x) {
        const int b = i / S, s = i - b * S;
        const float m = mask ? mask[(size_t)b * S + s] : 1.f;
        const float xm = (x[((size_t)b * C + c) * S + s] - mean) * m;
        loc += xm * xm;
    }
    sh[threadIdx.x] = loc; __syncthreads();
    for (int o = blockDim.x >> 1; o > 0; o >>= 1) {
        if (threadIdx.x < o) sh[threadIdx.x] += sh[threadIdx.x + o];
        __syncthreads();
    }
    if (threadIdx.x == 0) atomicAdd(&stats[513 + c], sh[0]);
}

__global__ void bn_apply_kernel(const float* __restrict__ x, const float* __restrict__ mask,
                                const float* __restrict__ stats, float* __restrict__ y,
                                int C, int S, int total)
{
    const int i = blockIdx.x * blockDim.x + threadIdx.x;
    if (i >= total) return;
    const int s = i % S; int t = i / S;
    const int c = t % C; const int b = t / C;
    const float cnt = fmaxf(stats[0], 1.f);
    const float mean = stats[1 + c] / cnt;
    const float var  = stats[513 + c] / cnt;
    const float m = mask ? mask[(size_t)b * S + s] : 1.f;
    const float v = (x[i] - mean) * m * rsqrtf(var + 1e-5f);
    y[i] = (v > 0.f) ? v : expm1f(v);
}

__global__ void pool2_kernel(const float* __restrict__ in, float* __restrict__ out,
                             int B, int Dout)
{
    const int S = Dout * Dout * Dout;
    const int i = blockIdx.x * blockDim.x + threadIdx.x;
    if (i >= B * S) return;
    const int s = i % S, b = i / S;
    const int ow = s % Dout; int t = s / Dout;
    const int oh = t % Dout; const int od = t / Dout;
    const int Din = 2 * Dout;
    const size_t base = (size_t)b * Din * Din * Din;
    float mx = -3.4e38f;
    for (int dz = 0; dz < 2; ++dz)
        for (int dy = 0; dy < 2; ++dy)
            for (int dx = 0; dx < 2; ++dx)
                mx = fmaxf(mx, in[base + (size_t)(2 * od + dz) * Din * Din
                                       + (2 * oh + dy) * Din + (2 * ow + dx)]);
    out[i] = mx;
}

__global__ void pool_big_kernel(const float* __restrict__ gt, const float* __restrict__ mask,
                                float* __restrict__ out, int G, int f)
{
    const int o = blockIdx.x;
    const int S = G * G * G;
    const int b = o / S, s = o % S;
    const int ow = s % G; int t = s / G;
    const int oh = t % G; const int od = t / G;
    const int Rr = 128;
    const size_t base = (size_t)b * Rr * Rr * Rr;
    const int f3 = f * f * f;
    float mx = -3.4e38f;
    for (int i = threadIdx.x; i < f3; i += blockDim.x) {
        const int dx = i % f; int tt = i / f;
        const int dy = tt % f; const int dz = tt / f;
        mx = fmaxf(mx, gt[base + (size_t)(od * f + dz) * Rr * Rr
                               + (oh * f + dy) * Rr + (ow * f + dx)]);
    }
    __shared__ float sh[256];
    sh[threadIdx.x] = mx; __syncthreads();
    for (int off = blockDim.x >> 1; off > 0; off >>= 1) {
        if (threadIdx.x < off) sh[threadIdx.x] = fmaxf(sh[threadIdx.x], sh[threadIdx.x + off]);
        __syncthreads();
    }
    if (threadIdx.x == 0) out[o] = sh[0] * (mask ? mask[o] : 1.f);
}

__global__ void gavg_kernel(const float* __restrict__ x, const float* __restrict__ mask,
                            float* __restrict__ g, int C, int S)
{
    const int bc = blockIdx.x;
    const int b = bc / C, c = bc % C;
    float sx = 0.f, sm = 0.f;
    for (int s = threadIdx.x; s < S; s += blockDim.x) {
        const float m = mask[(size_t)b * S + s];
        sx += x[((size_t)b * C + c) * S + s] * m;
        sm += m;
    }
    __shared__ float shx[256], shm[256];
    shx[threadIdx.x] = sx; shm[threadIdx.x] = sm; __syncthreads();
    for (int o = blockDim.x >> 1; o > 0; o >>= 1) {
        if (threadIdx.x < o) { shx[threadIdx.x] += shx[threadIdx.x + o]; shm[threadIdx.x] += shm[threadIdx.x + o]; }
        __syncthreads();
    }
    if (threadIdx.x == 0) g[bc] = shx[0] / fmaxf(shm[0], 1.f);
}

__global__ void linear_kernel(const float* __restrict__ g, const float* __restrict__ W,
                              const float* __restrict__ bias, float* __restrict__ out, int Cc)
{
    const int i = blockIdx.x * blockDim.x + threadIdx.x;
    if (i >= 2 * Cc) return;
    const int b = i / Cc, o = i % Cc;
    float acc = bias[o];
    for (int c = 0; c < Cc; ++c) acc += g[b * Cc + c] * W[c * Cc + o];
    out[i] = acc;
}

__global__ void zs_kernel(const float* __restrict__ means, const float* __restrict__ lv,
                          const float* __restrict__ eps, float* __restrict__ zws,
                          float* __restrict__ zout, int n)
{
    const int i = blockIdx.x * blockDim.x + threadIdx.x;
    if (i >= n) return;
    const float z = means[i] + expf(0.5f * lv[i]) * eps[i];
    zws[i] = z; zout[i] = z;
}

__global__ void tconv_kernel(const float* __restrict__ x, const float* __restrict__ w,
                             float* __restrict__ out, int B, int Cin, int Cout, int Din)
{
    const int Dout = 2 * Din;
    const int S = Dout * Dout * Dout;
    const int i = blockIdx.x * blockDim.x + threadIdx.x;
    if (i >= B * Cout * S) return;
    const int s = i % S; int t = i / S;
    const int co = t % Cout; const int b = t / Cout;
    const int ow = s % Dout; int t2 = s / Dout;
    const int oh = t2 % Dout; const int od = t2 / Dout;
    const int d = od >> 1, h = oh >> 1, wq = ow >> 1;
    const int tap = ((od & 1) << 2) | ((oh & 1) << 1) | (ow & 1);
    const int Sin = Din * Din * Din;
    float acc = 0.f;
    for (int ci = 0; ci < Cin; ++ci)
        acc += x[((size_t)b * Cin + ci) * Sin + d * Din * Din + h * Din + wq]
             * w[((size_t)ci * Cout + co) * 8 + tap];
    out[i] = acc;
}

__global__ void cls_kernel(const float* __restrict__ x, const float* __restrict__ w,
                           const float* __restrict__ bias, const float* __restrict__ mask,
                           float* __restrict__ out, int B, int Cc, int S)
{
    const int i = blockIdx.x * blockDim.x + threadIdx.x;
    if (i >= B * S) return;
    const int b = i / S, s = i % S;
    float acc = bias[0];
    for (int c = 0; c < Cc; ++c) acc += x[((size_t)b * Cc + c) * S + s] * w[c];
    out[i] = acc * (mask ? mask[i] : 1.f);
}

__global__ void keep_kernel(const float* __restrict__ cls, const float* __restrict__ tt,
                            const float* __restrict__ mask, float* __restrict__ keep, int n)
{
    const int i = blockIdx.x * blockDim.x + threadIdx.x;
    if (i >= n) return;
    const float m = mask ? mask[i] : 1.f;
    keep[i] = ((cls[i] > 0.f) || (tt[i] > 0.5f)) ? m : 0.f;
}

__global__ void mulmask_kernel(float* __restrict__ x, const float* __restrict__ mask,
                               int C, int S, int total)
{
    const int i = blockIdx.x * blockDim.x + threadIdx.x;
    if (i >= total) return;
    const int s = i % S; const int b = i / (C * S);
    x[i] *= mask[(size_t)b * S + s];
}

__global__ void mulmask_out_kernel(const float* __restrict__ x, const float* __restrict__ mask,
                                   float* __restrict__ out, int C, int S, int total)
{
    const int i = blockIdx.x * blockDim.x + threadIdx.x;
    if (i >= total) return;
    const int s = i % S; const int b = i / (C * S);
    out[i] = x[i] * mask[(size_t)b * S + s];
}

__global__ void up_kernel(const float* __restrict__ in, float* __restrict__ out, int B, int Din)
{
    const int Dout = 2 * Din;
    const int S = Dout * Dout * Dout;
    const int i = blockIdx.x * blockDim.x + threadIdx.x;
    if (i >= B * S) return;
    const int s = i % S, b = i / S;
    const int ow = s % Dout; int t = s / Dout;
    const int oh = t % Dout; const int od = t / Dout;
    out[i] = in[(size_t)b * Din * Din * Din + (od >> 1) * Din * Din
                + (oh >> 1) * Din + (ow >> 1)];
}

// ---------------------------------------------------------------------------
// Host-side helpers
// ---------------------------------------------------------------------------
static inline int imin(int a, int b) { return a < b ? a : b; }

static void launch_conv(hipStream_t st, const float* in, const float* w, float* out,
                        unsigned* bfU, unsigned* wpkU,
                        int Cin, int Cout, int B, int Din, int Hin, int Win,
                        int Dout, int Hout, int Wout, int stride)
{
    const int K = Cin * 27;
    const int K32 = (K + 31) & ~31;
    // activations -> bf16
    const int nIn = B * Cin * Din * Hin * Win;
    cvt_bf16_kernel<<<(nIn / 2 + 255) / 256, 256, 0, st>>>(in, bfU, nIn / 2);
    // weights -> packed bf16 [Cout][K32]
    const int pw = Cout * (K32 >> 1);
    pack_w_kernel<<<(pw + 255) / 256, 256, 0, st>>>(w, wpkU, Cout, K, K32);

    const int N = B * Dout * Hout * Wout;
    const int mt = (Cout >= 64) ? 4 : 1;
    const int nTiles = (N / 16) * (Cout / (16 * mt));
    const int blocks = (nTiles + 7) / 8;     // 8 wave32s per 256-thread block
    if (mt == 4)
        conv3_wmma_kernel<4><<<blocks, 256, 0, st>>>(
            (const unsigned short*)bfU, (const uint4*)wpkU, out,
            Cin, Cout, B, Din, Hin, Win, Dout, Hout, Wout, stride, K, K32, nTiles);
    else
        conv3_wmma_kernel<1><<<blocks, 256, 0, st>>>(
            (const unsigned short*)bfU, (const uint4*)wpkU, out,
            Cin, Cout, B, Din, Hin, Win, Dout, Hout, Wout, stride, K, K32, nTiles);
}

static void run_bn(hipStream_t st, float* x, const float* mask, float* stats,
                   int B, int Cc, int S)
{
    zero_kernel<<<(1025 + 255) / 256, 256, 0, st>>>(stats, 1025);
    const int BS = B * S;
    int gx = imin((BS + 1023) / 1024, 1024); if (gx < 1) gx = 1;
    dim3 g1((unsigned)gx, (unsigned)Cc);
    bn_sum_kernel<<<g1, 256, 0, st>>>(x, mask, stats, Cc, BS, S);
    bn_var_kernel<<<g1, 256, 0, st>>>(x, mask, stats, Cc, BS, S);
    const int total = B * Cc * S;
    bn_apply_kernel<<<(total + 255) / 256, 256, 0, st>>>(x, mask, stats, x, Cc, S, total);
}

// d_out layout (flat concat, reference return order)
#define OUT_CLS1   0
#define OUT_CLS2   128
#define OUT_T1     1152
#define OUT_T2     1280
#define OUT_X      2304
#define OUT_KEEP2  18688
#define OUT_MEANS  19712
#define OUT_LV     20736
#define OUT_ZS     21760

extern "C" void kernel_launch(void* const* d_in, const int* in_sizes, int n_in,
                              void* d_out_, int out_size, void* d_ws, size_t ws_size,
                              hipStream_t stream)
{
    (void)in_sizes; (void)n_in; (void)out_size; (void)ws_size;
    const float* occ    = (const float*)d_in[0];
    const float* gt     = (const float*)d_in[1];
    const float* eps    = (const float*)d_in[2];
    const float* e1c1   = (const float*)d_in[3];
    const float* e1c2   = (const float*)d_in[4];
    const float* e2c1   = (const float*)d_in[5];
    const float* e2c2   = (const float*)d_in[6];
    const float* e3c1   = (const float*)d_in[7];
    const float* e3c2   = (const float*)d_in[8];
    const float* wm     = (const float*)d_in[9];
    const float* bm     = (const float*)d_in[10];
    const float* wv     = (const float*)d_in[11];
    const float* bv     = (const float*)d_in[12];
    const float* d1t1   = (const float*)d_in[13];
    const float* d1c1   = (const float*)d_in[14];
    const float* d1t2   = (const float*)d_in[15];
    const float* d1c2   = (const float*)d_in[16];
    const float* cls1_w = (const float*)d_in[17];
    const float* cls1_b = (const float*)d_in[18];
    const float* d2t1   = (const float*)d_in[19];
    const float* d2c1   = (const float*)d_in[20];
    const float* cls2_w = (const float*)d_in[21];
    const float* cls2_b = (const float*)d_in[22];

    float* out = (float*)d_out_;
    float* ws  = (float*)d_ws;

    // workspace layout (floats); total ~25.1M floats (~100 MB)
    float* bufA  = ws;                 // 8388608 : (2,16,64^3) max fp32 tensor
    float* bufB  = bufA + 8388608;     // 8388608
    float* m1    = bufB + 8388608;     // 524288  : (2,1,64^3)
    float* m2    = m1 + 524288;        // 65536   : (2,1,32^3)
    float* m3    = m2 + 65536;         // 8192    : (2,1,16^3)
    float* stats = m3 + 8192;          // 2048
    float* g     = stats + 2048;       // 1024
    float* zsv   = g + 1024;           // 1024
    float* decA  = zsv + 1024;         // 16384
    float* decB  = decA + 16384;       // 16384
    float* keep1 = decB + 16384;       // 128
    float* mup8  = keep1 + 128;        // 1024
    unsigned* bfU  = (unsigned*)(mup8 + 1024);   // 4194304 u32 : bf16 activations
    unsigned* wpkU = bfU + 4194304;              // 3538944 u32 : packed bf16 weights

    // ---- masks (max-pool chain of occ) ----
    pool2_kernel<<<(2 * 64 * 64 * 64 + 255) / 256, 256, 0, stream>>>(occ, m1, 2, 64);
    pool2_kernel<<<(2 * 32 * 32 * 32 + 255) / 256, 256, 0, stream>>>(m1, m2, 2, 32);
    pool2_kernel<<<(2 * 16 * 16 * 16 + 255) / 256, 256, 0, stream>>>(m2, m3, 2, 16);

    // ---- encoder ----
    launch_conv(stream, occ,  e1c1, bufA, bfU, wpkU,   1,  16, 2, 128,128,128, 64,64,64, 2);
    run_bn(stream, bufA, m1, stats, 2, 16, 64*64*64);
    launch_conv(stream, bufA, e1c2, bufB, bfU, wpkU,  16,  16, 2,  64,64,64,  64,64,64, 1);
    run_bn(stream, bufB, m1, stats, 2, 16, 64*64*64);
    launch_conv(stream, bufB, e2c1, bufA, bfU, wpkU,  16,  64, 2,  64,64,64,  32,32,32, 2);
    run_bn(stream, bufA, m2, stats, 2, 64, 32*32*32);
    launch_conv(stream, bufA, e2c2, bufB, bfU, wpkU,  64,  64, 2,  32,32,32,  32,32,32, 1);
    run_bn(stream, bufB, m2, stats, 2, 64, 32*32*32);
    launch_conv(stream, bufB, e3c1, bufA, bfU, wpkU,  64, 512, 2,  32,32,32,  16,16,16, 2);
    run_bn(stream, bufA, m3, stats, 2, 512, 16*16*16);
    launch_conv(stream, bufA, e3c2, bufB, bfU, wpkU, 512, 512, 2,  16,16,16,  16,16,16, 1);
    run_bn(stream, bufB, m3, stats, 2, 512, 16*16*16);

    // ---- latent head ----
    gavg_kernel<<<2 * 512, 256, 0, stream>>>(bufB, m3, g, 512, 4096);
    linear_kernel<<<(1024 + 255) / 256, 256, 0, stream>>>(g, wm, bm, out + OUT_MEANS, 512);
    linear_kernel<<<(1024 + 255) / 256, 256, 0, stream>>>(g, wv, bv, out + OUT_LV, 512);
    zs_kernel<<<4, 256, 0, stream>>>(out + OUT_MEANS, out + OUT_LV, eps, zsv, out + OUT_ZS, 1024);

    // ---- decoder (masks all-ones until keep1) ----
    tconv_kernel<<<(2 * 512 * 8 + 255) / 256, 256, 0, stream>>>(zsv, d1t1, decA, 2, 512, 512, 1);
    run_bn(stream, decA, nullptr, stats, 2, 512, 8);
    launch_conv(stream, decA, d1c1, decB, bfU, wpkU, 512, 512, 2, 2,2,2, 2,2,2, 1);
    run_bn(stream, decB, nullptr, stats, 2, 512, 8);
    tconv_kernel<<<(2 * 64 * 64 + 255) / 256, 256, 0, stream>>>(decB, d1t2, decA, 2, 512, 64, 2);
    run_bn(stream, decA, nullptr, stats, 2, 64, 64);
    launch_conv(stream, decA, d1c2, decB, bfU, wpkU, 64, 64, 2, 4,4,4, 4,4,4, 1);
    run_bn(stream, decB, nullptr, stats, 2, 64, 64);

    cls_kernel<<<1, 128, 0, stream>>>(decB, cls1_w, cls1_b, nullptr, out + OUT_CLS1, 2, 64, 64);
    pool_big_kernel<<<128, 256, 0, stream>>>(gt, nullptr, out + OUT_T1, 4, 32);
    keep_kernel<<<1, 128, 0, stream>>>(out + OUT_CLS1, out + OUT_T1, nullptr, keep1, 128);
    mulmask_kernel<<<(2 * 64 * 64 + 255) / 256, 256, 0, stream>>>(decB, keep1, 64, 64, 2 * 64 * 64);
    up_kernel<<<(1024 + 255) / 256, 256, 0, stream>>>(keep1, mup8, 2, 4);

    tconv_kernel<<<(2 * 16 * 512 + 255) / 256, 256, 0, stream>>>(decB, d2t1, decA, 2, 64, 16, 4);
    run_bn(stream, decA, mup8, stats, 2, 16, 512);
    launch_conv(stream, decA, d2c1, decB, bfU, wpkU, 16, 16, 2, 8,8,8, 8,8,8, 1);
    run_bn(stream, decB, mup8, stats, 2, 16, 512);

    cls_kernel<<<(1024 + 255) / 256, 256, 0, stream>>>(decB, cls2_w, cls2_b, mup8, out + OUT_CLS2, 2, 16, 512);
    pool_big_kernel<<<1024, 256, 0, stream>>>(gt, mup8, out + OUT_T2, 8, 16);
    keep_kernel<<<4, 256, 0, stream>>>(out + OUT_CLS2, out + OUT_T2, mup8, out + OUT_KEEP2, 1024);
    mulmask_out_kernel<<<(16384 + 255) / 256, 256, 0, stream>>>(decB, out + OUT_KEEP2, out + OUT_X, 16, 512, 16384);
}
